// GCN_50208167690258
// MI455X (gfx1250) — matrix operand order repaired and checked by reading, not compiled, not run on hardware
//
#include <hip/hip_runtime.h>
#include <hip/hip_bf16.h>
#include <stdint.h>

#define NNODES 50000

typedef __attribute__((ext_vector_type(16))) __bf16 v16bf;
typedef __attribute__((ext_vector_type(8)))  __bf16 v8bf;
typedef __attribute__((ext_vector_type(4)))  __bf16 v4bf;
typedef __attribute__((ext_vector_type(8)))  float  v8f;

// ---------------- degree / normalization ----------------

__global__ void k_set_ones(float* __restrict__ deg, int n) {
    int i = blockIdx.x * blockDim.x + threadIdx.x;
    if (i < n) deg[i] = 1.0f;                 // self-loop contribution
}

__global__ void k_degree(const long long* __restrict__ dst, int E,
                         float* __restrict__ deg) {
    int e = blockIdx.x * blockDim.x + threadIdx.x;
    if (e < E) atomicAdd(&deg[(int)dst[e]], 1.0f);
}

__global__ void k_dinv(float* __restrict__ deg, int n) {
    int i = blockIdx.x * blockDim.x + threadIdx.x;
    if (i < n) deg[i] = rsqrtf(fmaxf(deg[i], 1.0f));   // in-place -> dinv
}

// ---------------- packing ----------------
// f32 -> bf16, 4 elements/thread (optionally fused ReLU for layer-2 input).

template<bool RELU>
__global__ void k_pack_bf16(const float* __restrict__ in,
                            __bf16* __restrict__ out, long long n4) {
    long long i = (long long)blockIdx.x * blockDim.x + threadIdx.x;
    if (i >= n4) return;
    float4 v = ((const float4*)in)[i];
    if (RELU) {
        v.x = fmaxf(v.x, 0.0f); v.y = fmaxf(v.y, 0.0f);
        v.z = fmaxf(v.z, 0.0f); v.w = fmaxf(v.w, 0.0f);
    }
    v4bf o = { (__bf16)v.x, (__bf16)v.y, (__bf16)v.z, (__bf16)v.w };
    *(v4bf*)(out + 4 * i) = o;
}

// W [128, NCOLS] f32 row-major -> WT [NCOLS, 128] bf16 (B fragments contiguous)
template<int NCOLS>
__global__ void k_pack_wt(const float* __restrict__ W, __bf16* __restrict__ WT) {
    int idx = blockIdx.x * blockDim.x + threadIdx.x;
    if (idx >= NCOLS * 128) return;
    int n = idx >> 7;           // output column
    int k = idx & 127;          // K index
    WT[idx] = (__bf16)W[(size_t)k * NCOLS + n];
}

// ---------------- fused WMMA GEMM + self-loop/bias epilogue ----------------
// Block = 8 waves. The block cooperatively stages WT (bf16, transposed) into
// LDS once; each wave then computes a 16 x NCOLS strip with all 4 A fragments
// (K=128) preloaded from global and B fragments served by ds_load_b128.
// Writes h = A@W (for the edge gather) and agg = h*dinv^2 + bias (self-loop
// message + bias pre-accumulated, so the scatter only adds edge messages).
// Fragment layouts per CDNA5 ISA 7.12.2 (wave32); the strip guard is
// wave-uniform (and placed after the barrier), so EXEC is all-1s at each WMMA.

template<int NCOLS>
__global__ void k_gemm_wmma_fused(const __bf16* __restrict__ A,
                                  const __bf16* __restrict__ WT,
                                  const float*  __restrict__ dinv,
                                  const float*  __restrict__ bias,
                                  float* __restrict__ h,
                                  float* __restrict__ agg, int M) {
    constexpr int NT = NCOLS / 16;
    __shared__ __bf16 sWT[NCOLS * 128];       // 32 KB (128) / 16 KB (64)

    // Cooperative stage of the whole weight tile (shared by the 8 waves).
    {
        constexpr int nvec = NCOLS * 128 / 8; // v8bf (16B) chunks
        for (int i = threadIdx.x; i < nvec; i += blockDim.x)
            ((v8bf*)sWT)[i] = ((const v8bf*)WT)[i];
    }
    __syncthreads();                          // all waves reach this barrier

    const int lane  = threadIdx.x & 31;
    const int strip = blockIdx.x * (blockDim.x >> 5) + (threadIdx.x >> 5);
    if (strip * 16 >= M) return;              // wave-uniform, after barrier
    const int half = lane >> 4;               // 0 or 1
    const int l15  = lane & 15;
    const int arow = strip * 16 + l15;

    // Preload all four A fragments (K = 0..127) in one burst.
    // Lanes 0-15 hold K=k0+[0..7] & k0+16+[0..7] of row M=lane;
    // lanes 16-31 hold K=k0+8+[0..7] & k0+24+[0..7].
    v16bf afrag[4];
    #pragma unroll
    for (int ks = 0; ks < 4; ++ks) {
        const __bf16* ap = A + (size_t)arow * 128 + ks * 32 + half * 8;
        v8bf alo = *(const v8bf*)ap;          // 16B contiguous -> b128
        v8bf ahi = *(const v8bf*)(ap + 16);
        afrag[ks] = __builtin_shufflevector(alo, ahi,
                      0,1,2,3,4,5,6,7,8,9,10,11,12,13,14,15);
    }

    v8f acc[NT];
    #pragma unroll
    for (int t = 0; t < NT; ++t) acc[t] = (v8f){};

    #pragma unroll
    for (int ks = 0; ks < 4; ++ks) {
        const int k0 = ks * 32;
        #pragma unroll
        for (int t = 0; t < NT; ++t) {
            // B fragment: lane n holds K=k0+half*8+[0..7] & k0+16+half*8+[0..7] of column n
            const __bf16* bp = sWT + (size_t)(t * 16 + l15) * 128 + k0 + half * 8;
            v8bf blo = *(const v8bf*)bp;      // ds_load_b128
            v8bf bhi = *(const v8bf*)(bp + 16);
            v16bf b = __builtin_shufflevector(blo, bhi,
                        0,1,2,3,4,5,6,7,8,9,10,11,12,13,14,15);
            acc[t] = __builtin_amdgcn_wmma_f32_16x16x32_bf16(
                /*neg_a=*/false, afrag[ks], /*neg_b=*/false, b,
                /*c_mod=*/(short)0, acc[t], /*reuse_a=*/false, /*reuse_b=*/false);
        }
    }

    // Epilogue: C/D layout VGPR r -> M = strip*16 + half*8 + r, N = col.
    float d2[8];
    #pragma unroll
    for (int r = 0; r < 8; ++r) {
        float di = dinv[strip * 16 + half * 8 + r];   // broadcast across 16 lanes
        d2[r] = di * di;
    }
    #pragma unroll
    for (int t = 0; t < NT; ++t) {
        const int col = t * 16 + l15;
        const float bc = bias[col];
        #pragma unroll
        for (int r = 0; r < 8; ++r) {
            size_t off = (size_t)(strip * 16 + half * 8 + r) * NCOLS + col;
            float v = acc[t][r];
            h[off]   = v;
            agg[off] = v * d2[r] + bc;
        }
    }
}

// ---------------- edge scatter ----------------
// NCOLS/4 threads per edge: float4 gather of h[src], scale by
// dinv[src]*dinv[dst], 4 fp32 global atomics into agg[dst].

template<int NCOLS>
__global__ void k_scatter(const long long* __restrict__ src,
                          const long long* __restrict__ dst, int E,
                          const float* __restrict__ h,
                          const float* __restrict__ dinv,
                          float* __restrict__ agg) {
    const int chunks = NCOLS / 4;
    long long gid = (long long)blockIdx.x * blockDim.x + threadIdx.x;
    long long e = gid / chunks;
    int c = (int)(gid - e * chunks);
    if (e >= E) return;
    int s = (int)src[e];
    int d = (int)dst[e];
    float w = dinv[s] * dinv[d];
    const float4 hv = *(const float4*)(h + (size_t)s * NCOLS + c * 4);
    float* ap = agg + (size_t)d * NCOLS + c * 4;
    atomicAdd(ap + 0, hv.x * w);
    atomicAdd(ap + 1, hv.y * w);
    atomicAdd(ap + 2, hv.z * w);
    atomicAdd(ap + 3, hv.w * w);
}

// ---------------- softmax over 64 features, one wave per node ----------------

__global__ void k_softmax64(const float* __restrict__ agg, float* __restrict__ out) {
    int lane = threadIdx.x & 31;
    int node = (int)(((long long)blockIdx.x * blockDim.x + threadIdx.x) >> 5);
    if (node >= NNODES) return;               // wave-uniform
    const float* r = agg + (size_t)node * 64;
    float v0 = r[lane], v1 = r[lane + 32];
    float m = fmaxf(v0, v1);
    #pragma unroll
    for (int off = 16; off > 0; off >>= 1)
        m = fmaxf(m, __shfl_xor(m, off, 32));
    float e0 = __expf(v0 - m), e1 = __expf(v1 - m);
    float sum = e0 + e1;
    #pragma unroll
    for (int off = 16; off > 0; off >>= 1)
        sum += __shfl_xor(sum, off, 32);
    float inv = __frcp_rn(sum);
    float* o = out + (size_t)node * 64;
    o[lane]      = e0 * inv;
    o[lane + 32] = e1 * inv;
}

// ---------------- launcher ----------------

extern "C" void kernel_launch(void* const* d_in, const int* in_sizes, int n_in,
                              void* d_out, int out_size, void* d_ws, size_t ws_size,
                              hipStream_t stream) {
    const float*     x  = (const float*)d_in[0];
    const long long* ei = (const long long*)d_in[1];   // int64 edge_index [2, E]
    const float*     W1 = (const float*)d_in[2];
    const float*     b1 = (const float*)d_in[3];
    const float*     W2 = (const float*)d_in[4];
    const float*     b2 = (const float*)d_in[5];
    float* out = (float*)d_out;

    const int E = in_sizes[1] / 2;
    const long long* src = ei;
    const long long* dst = ei + E;

    // ---- workspace carve (16B-aligned) ----
    char* p = (char*)d_ws;
    auto carve = [&](size_t bytes) {
        char* q = p;
        p += (bytes + 15) & ~(size_t)15;
        return q;
    };
    float*  dinv = (float*) carve((size_t)NNODES * 4);
    __bf16* wt1  = (__bf16*)carve((size_t)128 * 128 * 2);
    __bf16* wt2  = (__bf16*)carve((size_t)64 * 128 * 2);
    __bf16* xbf  = (__bf16*)carve((size_t)NNODES * 128 * 2);
    __bf16* r1bf = (__bf16*)carve((size_t)NNODES * 128 * 2);
    float*  h1   = (float*) carve((size_t)NNODES * 128 * 4);
    float*  agg1 = (float*) carve((size_t)NNODES * 128 * 4);
    float*  h2   = (float*) carve((size_t)NNODES * 64 * 4);
    float*  agg2 = (float*) carve((size_t)NNODES * 64 * 4);

    // ---- normalization coefficients ----
    k_set_ones<<<(NNODES + 255) / 256, 256, 0, stream>>>(dinv, NNODES);
    k_degree  <<<(E + 255) / 256,      256, 0, stream>>>(dst, E, dinv);
    k_dinv    <<<(NNODES + 255) / 256, 256, 0, stream>>>(dinv, NNODES);

    // ---- weight / input packing ----
    k_pack_wt<128><<<(128 * 128 + 255) / 256, 256, 0, stream>>>(W1, wt1);
    k_pack_wt<64> <<<(64 * 128 + 255) / 256,  256, 0, stream>>>(W2, wt2);
    {
        long long n4 = (long long)NNODES * 128 / 4;
        k_pack_bf16<false><<<(int)((n4 + 255) / 256), 256, 0, stream>>>(x, xbf, n4);
    }

    const int strips = NNODES / 16;           // 3125, exact

    // ---- layer 1: x @ W1 (+self-loop+bias fused) -> edge scatter ----
    k_gemm_wmma_fused<128><<<(strips + 7) / 8, 256, 0, stream>>>(
        xbf, wt1, dinv, b1, h1, agg1, NNODES);
    {
        long long work = (long long)E * (128 / 4);
        k_scatter<128><<<(int)((work + 255) / 256), 256, 0, stream>>>(
            src, dst, E, h1, dinv, agg1);
    }

    // ---- relu + pack layer-2 input ----
    {
        long long n4 = (long long)NNODES * 128 / 4;
        k_pack_bf16<true><<<(int)((n4 + 255) / 256), 256, 0, stream>>>(agg1, r1bf, n4);
    }

    // ---- layer 2: relu(agg1) @ W2 (+self-loop+bias fused) -> scatter -> softmax ----
    k_gemm_wmma_fused<64><<<(strips + 7) / 8, 256, 0, stream>>>(
        r1bf, wt2, dinv, b2, h2, agg2, NNODES);
    {
        long long work = (long long)E * (64 / 4);
        k_scatter<64><<<(int)((work + 255) / 256), 256, 0, stream>>>(
            src, dst, E, h2, dinv, agg2);
    }
    {
        long long sthreads = (long long)NNODES * 32;
        k_softmax64<<<(int)((sthreads + 255) / 256), 256, 0, stream>>>(agg2, out);
    }
}
